// ScaledDotProductAttention_45655502357195
// MI455X (gfx1250) — compile-verified, matrix operand order checked by reading, not run on hardware
//
#include <hip/hip_runtime.h>

typedef __attribute__((ext_vector_type(16))) _Float16 v16h;
typedef __attribute__((ext_vector_type(8)))  _Float16 v8h;
typedef __attribute__((ext_vector_type(8)))  float    v8f;

namespace {
constexpr int SEQ = 2048;
constexpr int DIM = 64;
constexpr int NH  = 16;
constexpr int NB  = 4;
constexpr float NEG_INF = -1000000000.0f;
constexpr float INV_KD  = 1.0f / 64.0f;   // reference divides by key_dim (not sqrt)
constexpr int QTILE = 16;
constexpr int NWAVE = 8;
constexpr int TILES_PER_WAVE = SEQ / 16 / NWAVE;  // 16 key-tiles per wave
}

__launch_bounds__(256, 1)
__global__ void attn_fwd_wmma(const float* __restrict__ Q,
                              const float* __restrict__ Kp,
                              const float* __restrict__ Vp,
                              const int*   __restrict__ Mask,
                              float* __restrict__ Out,
                              float* __restrict__ Attn)
{
  // 16 x 2048 f16 probability strip = 64 KB static LDS.
  // Aliased twice (barrier-protected):
  //   (a) float reduce buffer for softmax max/sum, dead before strip written
  //   (b) 4 KB of PV split-K partials, written after all strip reads complete
  __shared__ __align__(16) _Float16 sProbs[QTILE * SEQ];
  float* red = reinterpret_cast<float*>(sProbs);

  const int tid  = threadIdx.x;
  const int lane = tid & 31;
  const int wave = tid >> 5;
  const int qt = blockIdx.x;
  const int hh = blockIdx.y;
  const int bb = blockIdx.z;
  const int q0 = qt * QTILE;

  const size_t bh  = (size_t)bb * NH + hh;
  const size_t qkv = bh * SEQ * DIM;

  const int n    = lane & 15;   // A-frag row M / B-frag col N / C-frag col N
  const int half = lane >> 4;   // lane-half select
  const int moff = half * 8;    // C-frag row offset (M or M+8)
  const int kb   = half * 8;    // A/B-frag k sub-offset per ISA 16-bit layout

  // ---------------- A fragments from Q (row q0+n), pre-scaled by 1/64 --------
  v16h a0, a1;
  {
    const float* qrow = Q + qkv + (size_t)(q0 + n) * DIM;
#pragma unroll
    for (int c = 0; c < 2; ++c) {
      const float* p0 = qrow + c * 32 + kb;        // halves 0..7 : K = c*32+kb+i
      const float* p1 = qrow + c * 32 + 16 + kb;   // halves 8..15: K = c*32+16+kb+i
      float4 x0 = ((const float4*)p0)[0];
      float4 x1 = ((const float4*)p0)[1];
      float4 y0 = ((const float4*)p1)[0];
      float4 y1 = ((const float4*)p1)[1];
      float lo[8] = {x0.x, x0.y, x0.z, x0.w, x1.x, x1.y, x1.z, x1.w};
      float hi[8] = {y0.x, y0.y, y0.z, y0.w, y1.x, y1.y, y1.z, y1.w};
      v16h& a = c ? a1 : a0;
#pragma unroll
      for (int i = 0; i < 8; ++i) {
        a[i]     = (_Float16)(lo[i] * INV_KD);
        a[i + 8] = (_Float16)(hi[i] * INV_KD);
      }
    }
  }

  // ---------------- Phase 1: masked scores, kept entirely in registers -------
  v8f sc[TILES_PER_WAVE];
#pragma unroll
  for (int t = 0; t < TILES_PER_WAVE; ++t) {
    const int jc = (wave + t * NWAVE) * 16;
    const float* krow = Kp + qkv + (size_t)(jc + n) * DIM;  // B[k][n] = Key[n][k]
    if (t + 1 < TILES_PER_WAVE) {
      // overlap next tile's L2->WGP fill with this tile's cvt + WMMA
      const float* knext = Kp + qkv + (size_t)((wave + (t + 1) * NWAVE) * 16 + n) * DIM;
      __builtin_prefetch(knext, 0, 3);      // global_prefetch_b8
    }
    v16h b0, b1;
#pragma unroll
    for (int c = 0; c < 2; ++c) {
      const float* p0 = krow + c * 32 + kb;
      const float* p1 = krow + c * 32 + 16 + kb;
      float4 x0 = ((const float4*)p0)[0];
      float4 x1 = ((const float4*)p0)[1];
      float4 y0 = ((const float4*)p1)[0];
      float4 y1 = ((const float4*)p1)[1];
      float lo[8] = {x0.x, x0.y, x0.z, x0.w, x1.x, x1.y, x1.z, x1.w};
      float hi[8] = {y0.x, y0.y, y0.z, y0.w, y1.x, y1.y, y1.z, y1.w};
      v16h& bf = c ? b1 : b0;
#pragma unroll
      for (int i = 0; i < 8; ++i) {
        bf[i]     = (_Float16)lo[i];
        bf[i + 8] = (_Float16)hi[i];
      }
    }
    v8f c = {};
    c = __builtin_amdgcn_wmma_f32_16x16x32_f16(false, a0, false, b0, (short)0, c, false, false);
    c = __builtin_amdgcn_wmma_f32_16x16x32_f16(false, a1, false, b1, (short)0, c, false, false);
    const int* mrow = Mask + ((size_t)bb * SEQ + q0 + moff) * SEQ + jc + n;
#pragma unroll
    for (int g = 0; g < 8; ++g) {
      c[g] = (mrow[(size_t)g * SEQ] == 0) ? NEG_INF : c[g];
    }
    sc[t] = c;
  }

  // ---------------- row max: lane-local -> shfl_xor(1,2,4,8) -> cross-wave ---
  float mloc[8];
#pragma unroll
  for (int g = 0; g < 8; ++g) mloc[g] = -3.0e38f;
#pragma unroll
  for (int t = 0; t < TILES_PER_WAVE; ++t)
#pragma unroll
    for (int g = 0; g < 8; ++g) mloc[g] = fmaxf(mloc[g], sc[t][g]);
#pragma unroll
  for (int x = 1; x <= 8; x <<= 1)
#pragma unroll
    for (int g = 0; g < 8; ++g) mloc[g] = fmaxf(mloc[g], __shfl_xor(mloc[g], x, 32));
  if (n == 0) {   // lane 0 -> rows 0..7, lane 16 -> rows 8..15
#pragma unroll
    for (int g = 0; g < 8; ++g) red[wave * 16 + moff + g] = mloc[g];
  }
  __syncthreads();
  float rmax[8];
#pragma unroll
  for (int g = 0; g < 8; ++g) {
    float v = -3.0e38f;
#pragma unroll
    for (int w = 0; w < NWAVE; ++w) v = fmaxf(v, red[w * 16 + moff + g]);
    rmax[g] = v;
  }
  __syncthreads();

  // ---------------- exp + row sum (masked entries contribute 0) --------------
  float sloc[8];
#pragma unroll
  for (int g = 0; g < 8; ++g) sloc[g] = 0.0f;
#pragma unroll
  for (int t = 0; t < TILES_PER_WAVE; ++t)
#pragma unroll
    for (int g = 0; g < 8; ++g) {
      float s = sc[t][g];
      float p = (s == NEG_INF) ? 0.0f : __expf(s - rmax[g]);
      sc[t][g] = p;
      sloc[g] += p;
    }
#pragma unroll
  for (int x = 1; x <= 8; x <<= 1)
#pragma unroll
    for (int g = 0; g < 8; ++g) sloc[g] += __shfl_xor(sloc[g], x, 32);
  if (n == 0) {
#pragma unroll
    for (int g = 0; g < 8; ++g) red[wave * 16 + moff + g] = sloc[g];
  }
  __syncthreads();
  float rinv[8];
#pragma unroll
  for (int g = 0; g < 8; ++g) {
    float v = 0.0f;
#pragma unroll
    for (int w = 0; w < NWAVE; ++w) v += red[w * 16 + moff + g];
    rinv[g] = (v > 0.0f) ? (1.0f / v) : 0.0f;   // fully-masked row -> 0
  }
  __syncthreads();  // reduce buffer dead; strip may now be written

  // ---------------- normalize: attn to HBM (once) + f16 probs to LDS ---------
  float* attnBase = Attn + (bh * SEQ + (size_t)q0) * SEQ;
#pragma unroll
  for (int t = 0; t < TILES_PER_WAVE; ++t) {
    const int jc = (wave + t * NWAVE) * 16;
#pragma unroll
    for (int g = 0; g < 8; ++g) {
      float p = sc[t][g] * rinv[g];
      attnBase[(size_t)(moff + g) * SEQ + jc + n] = p;
      sProbs[(moff + g) * SEQ + jc + n] = (_Float16)p;
    }
  }
  __syncthreads();

  // ---------------- Phase 2: out = probs @ V, split-K over all 8 waves -------
  // wave w -> output N-tile (w & 3), K-half (w >> 2): 32 of 64 k-chunks each.
  const int nt    = wave & 3;
  const int khalf = wave >> 2;
  const int n0    = nt * 16;
  v8f acc = {};
  {
    const float* vcol = Vp + qkv + n0 + n;
#pragma unroll 1
    for (int kk = 0; kk < SEQ / 64; ++kk) {
      const int kc = khalf * (SEQ / 64) + kk;
      // A fragment: two contiguous 16B runs of f16 probs -> ds_load_b128
      const v8h* pa = (const v8h*)(sProbs + n * SEQ + kc * 32 + kb);
      const v8h* pb = (const v8h*)(sProbs + n * SEQ + kc * 32 + 16 + kb);
      v8h h0 = *pa, h1 = *pb;
      v16h af;
#pragma unroll
      for (int i = 0; i < 8; ++i) { af[i] = h0[i]; af[i + 8] = h1[i]; }
      // B fragment: V[k][n], k strided by DIM floats
      v16h bf;
      const float* vp0 = vcol + (size_t)(kc * 32 + kb) * DIM;
      const float* vp1 = vcol + (size_t)(kc * 32 + 16 + kb) * DIM;
#pragma unroll
      for (int i = 0; i < 8; ++i) {
        bf[i]     = (_Float16)vp0[(size_t)i * DIM];
        bf[i + 8] = (_Float16)vp1[(size_t)i * DIM];
      }
      acc = __builtin_amdgcn_wmma_f32_16x16x32_f16(false, af, false, bf, (short)0, acc, false, false);
    }
  }
  __syncthreads();  // all strip reads done; partials may alias strip start
  float* pred = reinterpret_cast<float*>(sProbs);
  if (khalf == 1) {
#pragma unroll
    for (int g = 0; g < 8; ++g) pred[nt * 256 + g * 32 + lane] = acc[g];
  }
  __syncthreads();
  if (khalf == 0) {
#pragma unroll
    for (int g = 0; g < 8; ++g) acc[g] += pred[nt * 256 + g * 32 + lane];
    float* orow = Out + qkv + (size_t)(q0 + moff) * DIM + n0 + n;
#pragma unroll
    for (int g = 0; g < 8; ++g) orow[(size_t)g * DIM] = acc[g];
  }
}

extern "C" void kernel_launch(void* const* d_in, const int* in_sizes, int n_in,
                              void* d_out, int out_size, void* d_ws, size_t ws_size,
                              hipStream_t stream) {
  const float* Q = (const float*)d_in[0];
  const float* K = (const float*)d_in[1];
  const float* V = (const float*)d_in[2];
  const int*   M = (const int*)d_in[3];
  float* Out  = (float*)d_out;
  float* Attn = Out + (size_t)NB * NH * SEQ * DIM;  // tuple: (out, attn) flat
  dim3 grid(SEQ / QTILE, NH, NB), block(256);
  hipLaunchKernelGGL(attn_fwd_wmma, grid, block, 0, stream, Q, K, V, M, Out, Attn);
}
